// WaveRNN_20882130993424
// MI455X (gfx1250) — compile-verified, matrix operand order checked by reading, not run on hardware
//
#include <hip/hip_runtime.h>
#include <hip/hip_bf16.h>

// ---------------------------------------------------------------------------
// Types for CDNA5 WMMA (gfx1250, wave32)
// ---------------------------------------------------------------------------
typedef __attribute__((ext_vector_type(16))) __bf16 v16bf;
typedef __attribute__((ext_vector_type(8)))  float  v8f;

#define H_DIM 512
#define E_DIM 512
#define N_MELS 80
#define T_STEPS 8192
#define BATCH 8
#define EPS_F 1e-5f

__device__ __forceinline__ unsigned short f2bf(float x) {
    unsigned int u = __float_as_uint(x);
    unsigned int r = u + 0x7FFFu + ((u >> 16) & 1u);   // round-to-nearest-even
    return (unsigned short)(r >> 16);
}

__device__ __forceinline__ v16bf ldbf16(const unsigned short* p) {
    return *reinterpret_cast<const v16bf*>(p);
}

// Fast gate activations on the v_exp/v_rcp transcendental pipe (no IEEE divide)
__device__ __forceinline__ float fast_sigmoid(float x) {
    return __builtin_amdgcn_rcpf(1.0f + __expf(-x));
}
__device__ __forceinline__ float fast_tanh(float x) {
    float e = __expf(-2.0f * x);
    return (1.0f - e) * __builtin_amdgcn_rcpf(1.0f + e);
}

// ===========================================================================
// Upsampler: per-output-channel block, fused conv + BN(stats+apply) + ReLU
// (+ optional residual).  in:(8,Lin,Ci)  w:(taps,Ci,512)  out:(8,Lout,512)
// ===========================================================================
__global__ __launch_bounds__(256)
void conv_bn_relu_kernel(const float* __restrict__ in, const float* __restrict__ w,
                         const float* __restrict__ bias, const float* __restrict__ bns,
                         const float* __restrict__ bno, const float* __restrict__ res,
                         float* __restrict__ out, int Lin, int Lout, int Ci, int taps,
                         int rate, int resOff)
{
    const int c   = blockIdx.x;          // 512 output channels
    const int tid = threadIdx.x;
    const int P   = BATCH * Lout;        // <= 304 positions
    float vals[2] = {0.f, 0.f};
    float lsum = 0.f, lsq = 0.f;
    #pragma unroll
    for (int s = 0; s < 2; ++s) {
        int p = tid + s * 256;
        if (p < P) {
            int b = p / Lout, t = p % Lout;
            float acc = bias[c];
            for (int tap = 0; tap < taps; ++tap) {
                const float* ip = in + ((size_t)b * Lin + (t + tap * rate)) * Ci;
                const float* wp = w + ((size_t)tap * Ci) * 512 + c;
                for (int ci = 0; ci < Ci; ++ci)
                    acc += ip[ci] * wp[(size_t)ci * 512];
            }
            vals[s] = acc; lsum += acc; lsq += acc * acc;
        }
    }
    __shared__ float s1[256], s2[256];
    s1[tid] = lsum; s2[tid] = lsq; __syncthreads();
    for (int st = 128; st > 0; st >>= 1) {
        if (tid < st) { s1[tid] += s1[tid + st]; s2[tid] += s2[tid + st]; }
        __syncthreads();
    }
    float mean = s1[0] / (float)P;
    float var  = s2[0] / (float)P - mean * mean;
    float scl  = bns[c] * rsqrtf(var + EPS_F);
    float off  = bno[c];
    #pragma unroll
    for (int s = 0; s < 2; ++s) {
        int p = tid + s * 256;
        if (p < P) {
            int b = p / Lout, t = p % Lout;
            float y = fmaxf((vals[s] - mean) * scl + off, 0.f);
            if (res) y += res[((size_t)b * Lin + (t + resOff)) * 512 + c];
            out[((size_t)b * Lout + t) * 512 + c] = y;
        }
    }
}

// Transposed conv (kernel 1, stride S) + BN + ReLU.  in:(8,Lin,512) -> (8,Lin*S,512)
__global__ __launch_bounds__(256)
void tconv_bn_relu_kernel(const float* __restrict__ in, const float* __restrict__ w,
                          const float* __restrict__ bias, const float* __restrict__ bns,
                          const float* __restrict__ bno, float* __restrict__ out,
                          int Lin, int stride)
{
    const int c   = blockIdx.x;
    const int tid = threadIdx.x;
    const int Lo  = Lin * stride;
    const int P   = BATCH * Lo;          // <= 4096
    float vals[16];
    float lsum = 0.f, lsq = 0.f;
    #pragma unroll
    for (int s = 0; s < 16; ++s) {
        int p = tid + s * 256;
        if (p < P) {
            int b = p / Lo, j = p % Lo;
            float acc = bias[c];
            if (j % stride == 0) {
                int t = j / stride;
                const float* ip = in + ((size_t)b * Lin + t) * 512;
                for (int ci = 0; ci < 512; ++ci)
                    acc += ip[ci] * w[(size_t)ci * 512 + c];
            }
            vals[s] = acc; lsum += acc; lsq += acc * acc;
        }
    }
    __shared__ float s1[256], s2[256];
    s1[tid] = lsum; s2[tid] = lsq; __syncthreads();
    for (int st = 128; st > 0; st >>= 1) {
        if (tid < st) { s1[tid] += s1[tid + st]; s2[tid] += s2[tid + st]; }
        __syncthreads();
    }
    float mean = s1[0] / (float)P;
    float var  = s2[0] / (float)P - mean * mean;
    float scl  = bns[c] * rsqrtf(var + EPS_F);
    float off  = bno[c];
    #pragma unroll
    for (int s = 0; s < 16; ++s) {
        int p = tid + s * 256;
        if (p < P) {
            int b = p / Lo, j = p % Lo;
            out[((size_t)b * Lo + j) * 512 + c] = fmaxf((vals[s] - mean) * scl + off, 0.f);
        }
    }
}

// ===========================================================================
// Swizzle packers into the CDNA5 16-bit WMMA A/B VGPR layouts.
// Element (m,k) of a 16x32 A tile lives at:
//   lane = m + 16*((k&15)>=8) ;  i = (k>=16 ? 8:0) + (k&7)   (i = vgpr*2+half)
// B (32x16) uses the same map with n in place of m.
// Layout in memory: [tile][kchunk][lane 0..31][i 0..15]  (32B per lane-chunk)
// ===========================================================================
__global__ __launch_bounds__(256)
void pack_b_kernel(const float* __restrict__ W, unsigned short* __restrict__ Bpk,
                   int Nt, int N)
{
    size_t idx = (size_t)blockIdx.x * 256 + threadIdx.x;
    size_t total = (size_t)Nt * 16 * 32 * 16;
    if (idx >= total) return;
    int i    = (int)(idx & 15);
    int lane = (int)((idx >> 4) & 31);
    int kc   = (int)((idx >> 9) & 15);
    int nt   = (int)(idx >> 13);
    int n = nt * 16 + (lane & 15);
    int k = kc * 32 + ((i >= 8) ? 16 : 0) + ((lane >= 16) ? 8 : 0) + (i & 7);
    Bpk[idx] = f2bf(W[(size_t)k * N + n]);
}

// A = embed_w[x] + cond_small[b][t>>4]   ->  bf16 swizzled (4096 M-tiles, K=512)
__global__ __launch_bounds__(256)
void pack_a_embed_kernel(const int* __restrict__ x, const float* __restrict__ embed_w,
                         const float* __restrict__ cond, unsigned short* __restrict__ Apk)
{
    size_t idx = (size_t)blockIdx.x * 256 + threadIdx.x;   // 4096*16*32*16 entries
    int i    = (int)(idx & 15);
    int lane = (int)((idx >> 4) & 31);
    int kc   = (int)((idx >> 9) & 15);
    int mt   = (int)(idx >> 13);
    int m = lane & 15;
    int r = mt * 16 + m;
    int b = r >> 13;              // r / 8192
    int t = r & 8191;
    int k = kc * 32 + ((i >= 8) ? 16 : 0) + ((lane >= 16) ? 8 : 0) + (i & 7);
    float val = embed_w[(size_t)x[(size_t)b * T_STEPS + t] * 512 + k]
              + cond[((size_t)(b * 512 + (t >> 4))) * 512 + k];
    Apk[idx] = f2bf(val);
}

// ===========================================================================
// Generic WMMA GEMM: wave computes one 16-row M tile x four 16-col N tiles.
// EPI 0: scatter fp32 to gx[t][b][n]   EPI 1: +bias,ReLU, re-swizzle bf16
// EPI 2: +bias, fp32 logits
// ===========================================================================
template<int EPI>
__global__ __launch_bounds__(256)
void wmma_gemm_kernel(const unsigned short* __restrict__ Apk,
                      const unsigned short* __restrict__ Bpk,
                      const float* __restrict__ bias,
                      float* __restrict__ outF,
                      unsigned short* __restrict__ outPk,
                      int NtDiv4, int Kc)
{
    const int w    = blockIdx.x * 8 + (threadIdx.x >> 5);
    const int lane = threadIdx.x & 31;
    const int mt   = w / NtDiv4;
    const int nt0  = (w % NtDiv4) * 4;

    v8f acc[4] = {};
    const unsigned short* ab = Apk + (((size_t)mt * Kc) * 32) * 16;
    #pragma unroll 1
    for (int kc = 0; kc < Kc; ++kc) {
        v16bf a = ldbf16(ab + (((size_t)kc * 32 + lane) << 4));
        #pragma unroll
        for (int j = 0; j < 4; ++j) {
            v16bf b = ldbf16(Bpk + ((((size_t)(nt0 + j) * Kc + kc) * 32 + lane) << 4));
            acc[j] = __builtin_amdgcn_wmma_f32_16x16x32_bf16(
                false, a, false, b, (short)0, acc[j], false, false);
        }
    }
    #pragma unroll
    for (int j = 0; j < 4; ++j) {
        int n = (nt0 + j) * 16 + (lane & 15);
        #pragma unroll
        for (int v = 0; v < 8; ++v) {
            int m = v + ((lane >> 4) << 3);
            int r = mt * 16 + m;
            float val = acc[j][v];
            if (EPI == 0) {                      // gx:  (t, b, 1536)
                int b = r >> 13, t = r & 8191;
                outF[((size_t)t * BATCH + b) * 1536 + n] = val;
            } else if (EPI == 1) {               // ReLU(o1) -> bf16 swizzled A
                val = fmaxf(val + bias[n], 0.f);
                int kc2 = n >> 5, kl = n & 31;
                int lane2 = m + (((kl & 15) >= 8) ? 16 : 0);
                int i2 = ((kl >= 16) ? 8 : 0) + (kl & 7);
                outPk[(((size_t)mt * 16 + kc2) * 32 + lane2) * 16 + i2] = f2bf(val);
            } else {                             // logits fp32
                outF[(size_t)r * 256 + n] = val + bias[n];
            }
        }
    }
}

// ===========================================================================
// Persistent sequential GRU scan.  One 512-thread (16-wave, wave32) workgroup.
// LDS: h fp32 (16x512) | A-tile swizzled bf16 (16KB) | gates 8x1536 | bias.
// The gate phase writes h_t straight into the swizzled LDS A-tile (rows 8..15
// stay zero), so there is no separate repack phase: 2 barriers per step.
// K-loop is 4-deep (4 independent accumulators) to keep 4 B-tile loads in
// flight against L2 latency.  Wh (bf16, 1.5MB) streams from L2 each step.
// ===========================================================================
__global__ __launch_bounds__(512, 1)
void gru_scan_kernel(const float* __restrict__ gx,
                     const unsigned short* __restrict__ bpk_wh,  // 96 N-tiles
                     const float* __restrict__ gru_b,
                     unsigned short* __restrict__ hpk)           // (4096,16,32,16)
{
    extern __shared__ char smem_raw[];
    float*          h_lds   = (float*)smem_raw;                       // 32 KB
    unsigned short* apk_lds = (unsigned short*)(smem_raw + 32768);    // 16 KB
    float*          hw_lds  = (float*)(smem_raw + 49152);             // 48 KB
    float*          b_lds   = (float*)(smem_raw + 98304);             // 6 KB

    const int tid  = threadIdx.x;
    const int wave = tid >> 5;       // 0..15
    const int lane = tid & 31;

    for (int idx = tid; idx < 16 * 512; idx += 512) h_lds[idx] = 0.f;
    for (int idx = tid; idx < 16 * 32 * 16; idx += 512) apk_lds[idx] = 0;
    for (int idx = tid; idx < 1536; idx += 512) b_lds[idx] = gru_b[idx];
    __syncthreads();

    for (int t = 0; t < T_STEPS; ++t) {
        // ---- phase 1: hw = h @ Wh  (16x1536); 6 N-tiles per wave, quad-acc K loop
        if (t + 1 < T_STEPS && tid < 384) {
            const char* nx = (const char*)(gx + ((size_t)(t + 1) * BATCH) * 1536);
            __builtin_prefetch(nx + tid * 128, 0, 1);
        }
        #pragma unroll 1
        for (int jj = 0; jj < 6; ++jj) {
            int nt = wave * 6 + jj;
            v8f acc0 = {}, acc1 = {}, acc2 = {}, acc3 = {};
            const unsigned short* bb = bpk_wh + (((size_t)nt * 16) * 32) * 16;
            #pragma unroll 1
            for (int kc = 0; kc < 16; kc += 4) {
                v16bf a0 = ldbf16(apk_lds + (((kc + 0) * 32 + lane) << 4));
                v16bf b0 = ldbf16(bb + (((size_t)(kc + 0) * 32 + lane) << 4));
                v16bf a1 = ldbf16(apk_lds + (((kc + 1) * 32 + lane) << 4));
                v16bf b1 = ldbf16(bb + (((size_t)(kc + 1) * 32 + lane) << 4));
                v16bf a2 = ldbf16(apk_lds + (((kc + 2) * 32 + lane) << 4));
                v16bf b2 = ldbf16(bb + (((size_t)(kc + 2) * 32 + lane) << 4));
                v16bf a3 = ldbf16(apk_lds + (((kc + 3) * 32 + lane) << 4));
                v16bf b3 = ldbf16(bb + (((size_t)(kc + 3) * 32 + lane) << 4));
                acc0 = __builtin_amdgcn_wmma_f32_16x16x32_bf16(
                    false, a0, false, b0, (short)0, acc0, false, false);
                acc1 = __builtin_amdgcn_wmma_f32_16x16x32_bf16(
                    false, a1, false, b1, (short)0, acc1, false, false);
                acc2 = __builtin_amdgcn_wmma_f32_16x16x32_bf16(
                    false, a2, false, b2, (short)0, acc2, false, false);
                acc3 = __builtin_amdgcn_wmma_f32_16x16x32_bf16(
                    false, a3, false, b3, (short)0, acc3, false, false);
            }
            if (lane < 16) {                     // rows 0..7 (real batch rows)
                v8f accA = acc0 + acc1;
                v8f accB = acc2 + acc3;
                #pragma unroll
                for (int v = 0; v < 8; ++v)
                    hw_lds[v * 1536 + nt * 16 + lane] = accA[v] + accB[v];
            }
        }
        __syncthreads();

        // ---- phase 2: gates + state update + write swizzled bf16 h_t
        const float* g0 = gx + ((size_t)t * BATCH) * 1536;
        #pragma unroll 1
        for (int idx = tid; idx < BATCH * 512; idx += 512) {
            int b = idx >> 9, j = idx & 511;
            const float* g  = g0 + (size_t)b * 1536;
            const float* hw = hw_lds + b * 1536;
            float z = fast_sigmoid(g[j]        + hw[j]        + b_lds[j]);
            float r = fast_sigmoid(g[512 + j]  + hw[512 + j]  + b_lds[512 + j]);
            float a = fast_tanh(   g[1024 + j] + r * hw[1024 + j] + b_lds[1024 + j]);
            float hn = (1.0f - z) * h_lds[b * 512 + j] + z * a;
            h_lds[b * 512 + j] = hn;
            unsigned short hb = f2bf(hn);
            // LDS A-tile for next step: element (m=b, k=j)
            int kl = j & 31;
            int laneA = b + (((kl & 15) >= 8) ? 16 : 0);
            int iA = ((kl >= 16) ? 8 : 0) + (kl & 7);
            apk_lds[(((j >> 5) * 32 + laneA) << 4) + iA] = hb;
            // hseq row = b*8192 + t  ->  Mtile = b*512 + (t>>4), m = t&15, k = j
            int mt = b * 512 + (t >> 4);
            int m  = t & 15;
            int lane2 = m + (((kl & 15) >= 8) ? 16 : 0);
            hpk[(((size_t)mt * 16 + (j >> 5)) * 32 + lane2) * 16 + iA] = hb;
        }
        __syncthreads();
    }
}

// ===========================================================================
// In-place log-softmax over the 256 classes (one block per row)
// ===========================================================================
__global__ __launch_bounds__(256)
void logsoftmax_kernel(float* __restrict__ out)
{
    const int row = blockIdx.x;
    const int tid = threadIdx.x;
    __shared__ float sm[256];
    float v = out[(size_t)row * 256 + tid];
    sm[tid] = v; __syncthreads();
    for (int s = 128; s > 0; s >>= 1) {
        if (tid < s) sm[tid] = fmaxf(sm[tid], sm[tid + s]);
        __syncthreads();
    }
    float mx = sm[0]; __syncthreads();
    float e = __expf(v - mx);
    sm[tid] = e; __syncthreads();
    for (int s = 128; s > 0; s >>= 1) {
        if (tid < s) sm[tid] += sm[tid + s];
        __syncthreads();
    }
    float lse = __logf(sm[0]);
    out[(size_t)row * 256 + tid] = v - mx - lse;
}

// ===========================================================================
// Host-side orchestration
// ===========================================================================
extern "C" void kernel_launch(void* const* d_in, const int* in_sizes, int n_in,
                              void* d_out, int out_size, void* d_ws, size_t ws_size,
                              hipStream_t stream)
{
    (void)in_sizes; (void)n_in; (void)out_size; (void)ws_size;
    const int*   x         = (const int*)  d_in[0];
    const float* mel       = (const float*)d_in[1];
    const float* embed_w   = (const float*)d_in[2];
    const float* conv_in_w = (const float*)d_in[3];
    const float* conv_in_b = (const float*)d_in[4];
    const float* conv_d1_w = (const float*)d_in[5];
    const float* conv_d1_b = (const float*)d_in[6];
    const float* conv_d2_w = (const float*)d_in[7];
    const float* conv_d2_b = (const float*)d_in[8];
    const float* up1_w     = (const float*)d_in[9];
    const float* up1_b     = (const float*)d_in[10];
    const float* up2_w     = (const float*)d_in[11];
    const float* up2_b     = (const float*)d_in[12];
    const float* up3_w     = (const float*)d_in[13];
    const float* up3_b     = (const float*)d_in[14];
    const float* bn_in_s   = (const float*)d_in[15];
    const float* bn_in_o   = (const float*)d_in[16];
    const float* bn_d1_s   = (const float*)d_in[17];
    const float* bn_d1_o   = (const float*)d_in[18];
    const float* bn_d2_s   = (const float*)d_in[19];
    const float* bn_d2_o   = (const float*)d_in[20];
    const float* bn_u1_s   = (const float*)d_in[21];
    const float* bn_u1_o   = (const float*)d_in[22];
    const float* bn_u2_s   = (const float*)d_in[23];
    const float* bn_u2_o   = (const float*)d_in[24];
    const float* bn_u3_s   = (const float*)d_in[25];
    const float* bn_u3_o   = (const float*)d_in[26];
    const float* gru_wi    = (const float*)d_in[27];
    const float* gru_wh    = (const float*)d_in[28];
    const float* gru_b     = (const float*)d_in[29];
    const float* o1_w      = (const float*)d_in[30];
    const float* o1_b      = (const float*)d_in[31];
    const float* o2_w      = (const float*)d_in[32];
    const float* o2_b      = (const float*)d_in[33];

    char* ws = (char*)d_ws;
    auto alloc = [&](size_t bytes) -> char* {
        char* p = ws;
        ws += (bytes + 255) & ~(size_t)255;
        return p;
    };
    float* u0   = (float*)alloc((size_t)8 * 38  * 512 * 4);
    float* u1   = (float*)alloc((size_t)8 * 36  * 512 * 4);
    float* u2   = (float*)alloc((size_t)8 * 32  * 512 * 4);
    float* t1   = (float*)alloc((size_t)8 * 64  * 512 * 4);
    float* t2   = (float*)alloc((size_t)8 * 128 * 512 * 4);
    float* cond = (float*)alloc((size_t)8 * 512 * 512 * 4);
    unsigned short* bpk_wi = (unsigned short*)alloc((size_t)96 * 16 * 32 * 16 * 2);
    unsigned short* bpk_wh = (unsigned short*)alloc((size_t)96 * 16 * 32 * 16 * 2);
    unsigned short* bpk_o1 = (unsigned short*)alloc((size_t)32 * 16 * 32 * 16 * 2);
    unsigned short* bpk_o2 = (unsigned short*)alloc((size_t)16 * 16 * 32 * 16 * 2);
    unsigned short* apk    = (unsigned short*)alloc((size_t)4096 * 16 * 32 * 16 * 2);
    unsigned short* hpk    = (unsigned short*)alloc((size_t)4096 * 16 * 32 * 16 * 2);
    unsigned short* h2pk   = (unsigned short*)alloc((size_t)4096 * 16 * 32 * 16 * 2);
    float* gx = (float*)alloc((size_t)T_STEPS * BATCH * 1536 * 4);

    float* logits = (float*)d_out;   // (8, 8192, 256)

    // ---- upsampler (fp32, fused conv+BN+ReLU per layer) ----
    conv_bn_relu_kernel<<<512, 256, 0, stream>>>(mel, conv_in_w, conv_in_b,
        bn_in_s, bn_in_o, nullptr, u0, 40, 38, 80, 3, 1, 0);
    conv_bn_relu_kernel<<<512, 256, 0, stream>>>(u0, conv_d1_w, conv_d1_b,
        bn_d1_s, bn_d1_o, u0, u1, 38, 36, 512, 2, 2, 1);
    conv_bn_relu_kernel<<<512, 256, 0, stream>>>(u1, conv_d2_w, conv_d2_b,
        bn_d2_s, bn_d2_o, u1, u2, 36, 32, 512, 2, 4, 2);
    tconv_bn_relu_kernel<<<512, 256, 0, stream>>>(u2, up1_w, up1_b,
        bn_u1_s, bn_u1_o, t1, 32, 2);
    tconv_bn_relu_kernel<<<512, 256, 0, stream>>>(t1, up2_w, up2_b,
        bn_u2_s, bn_u2_o, t2, 64, 2);
    tconv_bn_relu_kernel<<<512, 256, 0, stream>>>(t2, up3_w, up3_b,
        bn_u3_s, bn_u3_o, cond, 128, 4);

    // ---- bf16 swizzle packs ----
    pack_b_kernel<<<(96 * 16 * 32 * 16) / 256, 256, 0, stream>>>(gru_wi, bpk_wi, 96, 1536);
    pack_b_kernel<<<(96 * 16 * 32 * 16) / 256, 256, 0, stream>>>(gru_wh, bpk_wh, 96, 1536);
    pack_b_kernel<<<(32 * 16 * 32 * 16) / 256, 256, 0, stream>>>(o1_w,  bpk_o1, 32, 512);
    pack_b_kernel<<<(16 * 16 * 32 * 16) / 256, 256, 0, stream>>>(o2_w,  bpk_o2, 16, 256);
    pack_a_embed_kernel<<<(4096 * 16 * 32 * 16) / 256, 256, 0, stream>>>(x, embed_w, cond, apk);

    // ---- gx = inp @ Wi  (65536 x 512 x 1536, bf16 WMMA) ----
    wmma_gemm_kernel<0><<<(4096 * (96 / 4)) / 8, 256, 0, stream>>>(
        apk, bpk_wi, nullptr, gx, nullptr, 96 / 4, 16);

    // ---- sequential GRU (persistent single workgroup, ~102 KB LDS) ----
    gru_scan_kernel<<<1, 512, 104448, stream>>>(gx, bpk_wh, gru_b, hpk);

    // ---- output head ----
    wmma_gemm_kernel<1><<<(4096 * (32 / 4)) / 8, 256, 0, stream>>>(
        hpk, bpk_o1, o1_b, nullptr, h2pk, 32 / 4, 16);
    wmma_gemm_kernel<2><<<(4096 * (16 / 4)) / 8, 256, 0, stream>>>(
        h2pk, bpk_o2, o2_b, logits, nullptr, 16 / 4, 16);
    logsoftmax_kernel<<<BATCH * T_STEPS, 256, 0, stream>>>(logits);
}
